// HeteroAttention_20899310863111
// MI455X (gfx1250) — compile-verified, matrix operand order checked by reading, not run on hardware
//
#include <hip/hip_runtime.h>
#include <math.h>

#define NN   50000
#define EE   100000
#define HH   8
#define DHH  32
#define HIDD 256
#define QKSCALE 0.17677669529663687f  // 1/sqrt(32)

typedef __attribute__((ext_vector_type(16))) __bf16 v16bf;
typedef __attribute__((ext_vector_type(8)))  __bf16 v8bf;
typedef __attribute__((ext_vector_type(8)))  float  v8f;

// ---------------------------------------------------------------- helpers
__device__ __forceinline__ __bf16 f2bf(float f) {
  unsigned int u = __float_as_uint(f);
  u = u + 0x7FFFu + ((u >> 16) & 1u);         // round-to-nearest-even
  unsigned short s = (unsigned short)(u >> 16);
  __bf16 b;
  __builtin_memcpy(&b, &s, 2);
  return b;
}

__device__ __forceinline__ void atomicMaxFloat(float* addr, float val) {
  if (!(val == val)) return;
  if (val >= 0.0f) atomicMax((int*)addr, __float_as_int(val));
  else             atomicMin((unsigned int*)addr, (unsigned int)__float_as_int(val));
}

// ---------------------------------------------------------------- utility kernels
__global__ void fill_kernel(float* __restrict__ p, float v, long long n) {
  long long i = (long long)blockIdx.x * blockDim.x + threadIdx.x;
  long long stride = (long long)gridDim.x * blockDim.x;
  for (; i < n; i += stride) p[i] = v;
}

__global__ void f32_to_bf16_kernel(const float* __restrict__ in, __bf16* __restrict__ out,
                                   long long n) {
  long long i = (long long)blockIdx.x * blockDim.x + threadIdx.x;
  long long stride = (long long)gridDim.x * blockDim.x;
  for (; i < n; i += stride) out[i] = f2bf(in[i]);
}

// W: [256,256] row-major (W[k*256+n]); Wt: [n*256+k] bf16 (transposed, row-contiguous in k)
__global__ void wtrans_kernel(const float* __restrict__ W, __bf16* __restrict__ Wt) {
  int i = blockIdx.x * 256 + threadIdx.x;   // 65536 elements
  int n = i >> 8, k = i & 255;
  Wt[i] = f2bf(W[k * 256 + n]);
}

// ---------------------------------------------------------------- WMMA GEMM
// Y[M,256] = Xbf[M,256] @ W + bias, with W supplied transposed as Wt[n*256+k] (bf16).
// One wave computes a 16(M) x 64(N) strip: the A fragment is loaded once per k-step
// and reused across 4 WMMAs (4 N-tiles), so 10 b128 loads feed 4 matrix ops.
// K=256 fully unrolled -> 32x v_wmma_f32_16x16x32_bf16 per wave.
// A-fragment lane layout (16-bit A 16x32): lanes 0-15 hold K={0..7,16..23},
// lanes 16-31 hold K={8..15,24..31}; B mirrored with column in lane.
__global__ __launch_bounds__(32)
void gemm_bias_wmma(const __bf16* __restrict__ X, const __bf16* __restrict__ Wt,
                    const float* __restrict__ bias, float* __restrict__ Y) {
  const int lane = threadIdx.x & 31;
  const int lo = lane & 15, hi = lane >> 4;
  const int tm  = blockIdx.x * 16;
  const int tn0 = blockIdx.y * 64;

  const __bf16* arow = X  + (size_t)(tm  + lo) * HIDD;
  const __bf16* brow = Wt + (size_t)(tn0 + lo) * HIDD;   // N-tile t adds t*16*HIDD

  v8f acc[4] = {{}, {}, {}, {}};
#pragma unroll
  for (int kb = 0; kb < HIDD; kb += 32) {
    v8bf a0 = *(const v8bf*)(arow + kb + hi * 8);
    v8bf a1 = *(const v8bf*)(arow + kb + 16 + hi * 8);
    v16bf a = __builtin_shufflevector(a0, a1, 0,1,2,3,4,5,6,7,8,9,10,11,12,13,14,15);
#pragma unroll
    for (int t = 0; t < 4; ++t) {
      const __bf16* bt = brow + (size_t)t * 16 * HIDD;
      v8bf b0 = *(const v8bf*)(bt + kb + hi * 8);
      v8bf b1 = *(const v8bf*)(bt + kb + 16 + hi * 8);
      v16bf b = __builtin_shufflevector(b0, b1, 0,1,2,3,4,5,6,7,8,9,10,11,12,13,14,15);
      acc[t] = __builtin_amdgcn_wmma_f32_16x16x32_bf16(false, a, false, b,
                                                       (short)0, acc[t], false, false);
    }
  }

#pragma unroll
  for (int t = 0; t < 4; ++t) {
    const float bv = bias[tn0 + t * 16 + lo];
#pragma unroll
    for (int j = 0; j < 8; ++j) {
      int m = hi * 8 + j;                       // C/D layout: vgpr j, lanes16-31 -> M+8
      Y[(size_t)(tm + m) * HIDD + tn0 + t * 16 + lo] = acc[t][j] + bv;
    }
  }
}

// ---------------------------------------------------------------- edge attention
// One 32-lane group per (edge, head); lane = DH element.
__global__ __launch_bounds__(256)
void scores_kernel(const float* __restrict__ q, const float* __restrict__ k,
                   const float* __restrict__ ee, const int* __restrict__ ei,
                   const float* __restrict__ ctw, float* __restrict__ s) {
  int g = (int)(((long long)blockIdx.x * blockDim.x + threadIdx.x) >> 5);
  int lane = threadIdx.x & 31;
  if (g >= EE * HH) return;
  int e = g >> 3, h = g & 7;
  int r = ei[e], c = ei[EE + e];
  float qv = q[((size_t)c * HH + h) * DHH + lane];
  float kv = k[((size_t)r * HH + h) * DHH + lane];
  float ev = ee[((size_t)e * HH + h) * DHH + lane];
  float p = qv * kv * QKSCALE + qv * ev;
#pragma unroll
  for (int off = 16; off; off >>= 1) p += __shfl_xor(p, off, 32);
  if (lane == 0) s[(size_t)e * HH + h] = p + ctw[h];
}

// Per-head max over all edges (global softmax axis=0). Thread t owns head t&7.
__global__ __launch_bounds__(256)
void head_max_kernel(const float* __restrict__ s, float* __restrict__ headmax, int n) {
  __shared__ float sm[256];
  int t = threadIdx.x, h = t & 7;
  float m = -INFINITY;
  int chunk = blockIdx.x * 32 + (t >> 3);
  int cstride = gridDim.x * 32;
  for (int i = chunk * 8 + h; i < n; i += cstride * 8) m = fmaxf(m, s[i]);
  sm[t] = m; __syncthreads();
  for (int off = 128; off >= 8; off >>= 1) {
    if (t < off) sm[t] = fmaxf(sm[t], sm[t + off]);
    __syncthreads();
  }
  if (t < 8) atomicMaxFloat(&headmax[t], sm[t]);
}

// In-place exp(s - max[h]) and per-head sum accumulation.
__global__ __launch_bounds__(256)
void exp_sum_kernel(float* __restrict__ s, const float* __restrict__ headmax,
                    float* __restrict__ headsum, int n) {
  __shared__ float sm[256];
  int t = threadIdx.x, h = t & 7;
  float mx = headmax[h];
  float acc = 0.0f;
  int chunk = blockIdx.x * 32 + (t >> 3);
  int cstride = gridDim.x * 32;
  for (int i = chunk * 8 + h; i < n; i += cstride * 8) {
    float ev = expf(s[i] - mx);
    s[i] = ev;
    acc += ev;
  }
  sm[t] = acc; __syncthreads();
  for (int off = 128; off >= 8; off >>= 1) {
    if (t < off) sm[t] += sm[t + off];
    __syncthreads();
  }
  if (t < 8) atomicAdd(&headsum[t], sm[t]);
}

// attn-weighted scatter of v into att[dst], plus head-mean totals.
__global__ __launch_bounds__(256)
void scatter_kernel(const float* __restrict__ attn, const float* __restrict__ headsum,
                    const float* __restrict__ v, const int* __restrict__ ei,
                    float* __restrict__ att, float* __restrict__ tot) {
  int g = (int)(((long long)blockIdx.x * blockDim.x + threadIdx.x) >> 5);
  int lane = threadIdx.x & 31;
  if (g >= EE * HH) return;
  int e = g >> 3, h = g & 7;
  int r = ei[e], c = ei[EE + e];
  float w = attn[(size_t)e * HH + h] / headsum[h];
  float vv = v[((size_t)r * HH + h) * DHH + lane];
  atomicAdd(&att[((size_t)c * HH + h) * DHH + lane], w * vv);
  if (lane == 0) atomicAdd(&tot[c], w * (1.0f / HH));
}

// agg = att / clip(tot, 1e-8) -> bf16 for the output GEMM
__global__ __launch_bounds__(256)
void norm_bf16_kernel(const float* __restrict__ att, const float* __restrict__ tot,
                      __bf16* __restrict__ out, long long n) {
  long long i = (long long)blockIdx.x * blockDim.x + threadIdx.x;
  if (i >= n) return;
  float t = fmaxf(tot[i >> 8], 1e-8f);
  out[i] = f2bf(att[i] / t);
}

// ---------------------------------------------------------------- host driver
extern "C" void kernel_launch(void* const* d_in, const int* in_sizes, int n_in,
                              void* d_out, int out_size, void* d_ws, size_t ws_size,
                              hipStream_t stream) {
  (void)in_sizes; (void)n_in; (void)out_size; (void)ws_size;

  const float* x_txn  = (const float*)d_in[0];
  const float* x_acct = (const float*)d_in[1];
  const float* ea[3]  = {(const float*)d_in[2], (const float*)d_in[3], (const float*)d_in[4]};
  const float* Ws[10] = {(const float*)d_in[5],  (const float*)d_in[7],  (const float*)d_in[9],
                         (const float*)d_in[11], (const float*)d_in[13], (const float*)d_in[15],
                         (const float*)d_in[17], (const float*)d_in[19], (const float*)d_in[21],
                         (const float*)d_in[26]};
  const float* bq_txn  = (const float*)d_in[6];
  const float* bk_txn  = (const float*)d_in[8];
  const float* bv_txn  = (const float*)d_in[10];
  const float* bq_acct = (const float*)d_in[12];
  const float* bk_acct = (const float*)d_in[14];
  const float* bv_acct = (const float*)d_in[16];
  const float* be[3]   = {(const float*)d_in[18], (const float*)d_in[20], (const float*)d_in[22]};
  const float* ctw[3]  = {(const float*)d_in[23], (const float*)d_in[24], (const float*)d_in[25]};
  const float* bo      = (const float*)d_in[27];
  const int*   ei[3]   = {(const int*)d_in[28], (const int*)d_in[29], (const int*)d_in[30]};

  // ---- carve workspace (256B aligned)
  char* base = (char*)d_ws;
  size_t off = 0;
  auto alloc = [&](size_t bytes) -> void* {
    off = (off + 255) & ~(size_t)255;
    void* p = base + off;
    off += bytes;
    return p;
  };
  const size_t NH = (size_t)NN * HIDD, EH = (size_t)EE * HIDD;
  float* q_txn   = (float*)alloc(NH * 4);
  float* k_txn   = (float*)alloc(NH * 4);
  float* v_txn   = (float*)alloc(NH * 4);
  float* q_acct  = (float*)alloc(NH * 4);
  float* k_acct  = (float*)alloc(NH * 4);
  float* v_acct  = (float*)alloc(NH * 4);
  float* eebuf   = (float*)alloc(EH * 4);
  float* att_txn = (float*)alloc(NH * 4);
  float* att_acct= (float*)alloc(NH * 4);
  float* tot_txn = (float*)alloc((size_t)NN * 4);
  float* tot_acct= (float*)alloc((size_t)NN * 4);
  float* scores  = (float*)alloc((size_t)EE * HH * 4);
  float* headmax = (float*)alloc(8 * 4);
  float* headsum = (float*)alloc(8 * 4);
  __bf16* xb_txn  = (__bf16*)alloc(NH * 2);
  __bf16* xb_acct = (__bf16*)alloc(NH * 2);
  __bf16* eab     = (__bf16*)alloc(EH * 2);
  __bf16* aggb    = (__bf16*)alloc(NH * 2);
  __bf16* Wtb     = (__bf16*)alloc((size_t)10 * 65536 * 2);

  // ---- stage weights (transposed bf16) and activations (bf16)
  for (int i = 0; i < 10; ++i)
    wtrans_kernel<<<256, 256, 0, stream>>>(Ws[i], Wtb + (size_t)i * 65536);
  f32_to_bf16_kernel<<<4096, 256, 0, stream>>>(x_txn,  xb_txn,  (long long)NH);
  f32_to_bf16_kernel<<<4096, 256, 0, stream>>>(x_acct, xb_acct, (long long)NH);

  // ---- QKV projections (WMMA): one wave per 16x64 strip
  dim3 gN(NN / 16, HIDD / 64), gE(EE / 16, HIDD / 64);
  gemm_bias_wmma<<<gN, 32, 0, stream>>>(xb_txn,  Wtb + 0 * 65536, bq_txn,  q_txn);
  gemm_bias_wmma<<<gN, 32, 0, stream>>>(xb_txn,  Wtb + 1 * 65536, bk_txn,  k_txn);
  gemm_bias_wmma<<<gN, 32, 0, stream>>>(xb_txn,  Wtb + 2 * 65536, bv_txn,  v_txn);
  gemm_bias_wmma<<<gN, 32, 0, stream>>>(xb_acct, Wtb + 3 * 65536, bq_acct, q_acct);
  gemm_bias_wmma<<<gN, 32, 0, stream>>>(xb_acct, Wtb + 4 * 65536, bk_acct, k_acct);
  gemm_bias_wmma<<<gN, 32, 0, stream>>>(xb_acct, Wtb + 5 * 65536, bv_acct, v_acct);

  // ---- zero accumulators
  fill_kernel<<<2048, 256, 0, stream>>>(att_txn,  0.0f, (long long)NH);
  fill_kernel<<<2048, 256, 0, stream>>>(att_acct, 0.0f, (long long)NH);
  fill_kernel<<<64,   256, 0, stream>>>(tot_txn,  0.0f, NN);
  fill_kernel<<<64,   256, 0, stream>>>(tot_acct, 0.0f, NN);

  // ---- per-relation edge attention (tt, at, ta)
  const float* qp[3] = {q_txn, q_txn, q_acct};
  const float* kp[3] = {k_txn, k_acct, k_txn};
  const float* vp[3] = {v_txn, v_acct, v_txn};
  float* attp[3] = {att_txn, att_txn, att_acct};
  float* totp[3] = {tot_txn, tot_txn, tot_acct};

  const int egroups = (EE * HH) / 8;   // 32-lane groups, 8 per 256-thread block
  for (int rel = 0; rel < 3; ++rel) {
    f32_to_bf16_kernel<<<4096, 256, 0, stream>>>(ea[rel], eab, (long long)EH);
    gemm_bias_wmma<<<gE, 32, 0, stream>>>(eab, Wtb + (size_t)(6 + rel) * 65536,
                                          be[rel], eebuf);
    scores_kernel<<<egroups, 256, 0, stream>>>(qp[rel], kp[rel], eebuf, ei[rel],
                                               ctw[rel], scores);
    fill_kernel<<<1, 32, 0, stream>>>(headmax, -INFINITY, 8);
    fill_kernel<<<1, 32, 0, stream>>>(headsum, 0.0f, 8);
    head_max_kernel<<<256, 256, 0, stream>>>(scores, headmax, EE * HH);
    exp_sum_kernel<<<256, 256, 0, stream>>>(scores, headmax, headsum, EE * HH);
    scatter_kernel<<<egroups, 256, 0, stream>>>(scores, headsum, vp[rel], ei[rel],
                                                attp[rel], totp[rel]);
  }

  // ---- normalize + output projection (WMMA) straight into d_out
  float* out_txn  = (float*)d_out;
  float* out_acct = (float*)d_out + NH;
  norm_bf16_kernel<<<(unsigned)(NH / 256), 256, 0, stream>>>(att_txn, tot_txn, aggb,
                                                             (long long)NH);
  gemm_bias_wmma<<<gN, 32, 0, stream>>>(aggb, Wtb + 9 * 65536, bo, out_txn);
  norm_bf16_kernel<<<(unsigned)(NH / 256), 256, 0, stream>>>(att_acct, tot_acct, aggb,
                                                             (long long)NH);
  gemm_bias_wmma<<<gN, 32, 0, stream>>>(aggb, Wtb + 9 * 65536, bo, out_acct);
}
